// DNALayer_25898652795052
// MI455X (gfx1250) — matrix operand order repaired and checked by reading, not compiled, not run on hardware
//
#include <hip/hip_runtime.h>
#include <math.h>

// ---------------- problem constants ----------------
#define BB 2
#define LL 2048
#define EE 1024
#define HH 16
#define DD 64
#define KSW 13
#define NCHUNK 8                       // L-chunking for ws reuse
#define GPC (BB*LL/NCHUNK)             // (b,l) rows per chunk = 512
#define MPC (GPC*KSW)                  // sampled rows per chunk = 6656
#define NTBL_C ((size_t)BB*LL*KSW*HH)  // 851,968 table entries

typedef __attribute__((ext_vector_type(2)))  float    v2f;
typedef __attribute__((ext_vector_type(8)))  float    v8f;
typedef __attribute__((ext_vector_type(16))) _Float16 v16h;
typedef __attribute__((ext_vector_type(4)))  unsigned u32x4;
typedef __attribute__((ext_vector_type(4)))  int      i32x4;
typedef __attribute__((ext_vector_type(8)))  int      i32x8;

// V_WMMA_F32_16X16X4_F32 (exact fp32 path, Q/O projections)
__device__ __forceinline__ v8f wmma_f32(v2f a, v2f b, v8f c) {
  return __builtin_amdgcn_wmma_f32_16x16x4_f32(
      false, a, false, b, (short)0, c, false, false);
}
// V_WMMA_F32_16X16X32_F16 (high-throughput pipe, K/V projections)
__device__ __forceinline__ v8f wmma_f16(v16h a, v16h b, v8f c) {
  return __builtin_amdgcn_wmma_f32_16x16x32_f16(
      false, a, false, b, (short)0, c, false, false);
}

// TDM: 2D tile (tile_dim0 elems x tile_dim1 rows, 4-byte elems) global -> LDS
__device__ __forceinline__ void tdm_load_2d_b32(unsigned lds_off,
                                                const void* gaddr,
                                                unsigned tile_d0, unsigned tile_d1,
                                                unsigned long long row_stride_elems)
{
  unsigned long long ga = (unsigned long long)(uintptr_t)gaddr;
  u32x4 g0;
  g0[0] = 1u;                                            // count=1, user descriptor
  g0[1] = lds_off;                                       // LDS byte address
  g0[2] = (unsigned)ga;                                  // global addr [31:0]
  g0[3] = (unsigned)((ga >> 32) & 0x1FFFFFFu) | (2u << 30); // addr[56:32] | type=2
  i32x8 g1;
  g1[0] = 0x20000;                                       // data_size=2 (4B), mask=0
  g1[1] = (int)((tile_d0 & 0xFFFFu) << 16);              // tensor_dim0[15:0]
  g1[2] = (int)((tile_d0 >> 16) | ((tile_d1 & 0xFFFFu) << 16)); // dim0[31:16]|dim1[15:0]
  g1[3] = (int)((tile_d1 >> 16) | (tile_d0 << 16));      // dim1[31:16] | tile_dim0
  g1[4] = (int)(tile_d1 & 0xFFFFu);                      // tile_dim1 (tile_dim2=0)
  g1[5] = (int)(unsigned)(row_stride_elems & 0xFFFFFFFFull); // dim0_stride[31:0]
  g1[6] = (int)(unsigned)(row_stride_elems >> 32);       // dim0_stride[47:32]
  g1[7] = 0;
  i32x4 gz = {0, 0, 0, 0};
#if __clang_major__ >= 23
  i32x8 gz8 = {0, 0, 0, 0, 0, 0, 0, 0};
  __builtin_amdgcn_tensor_load_to_lds(g0, g1, gz, gz, gz8, 0);
#else
  __builtin_amdgcn_tensor_load_to_lds(g0, g1, gz, gz, 0);
#endif
}

// ---------------------------------------------------------------------------
// Weight split: W(f32) -> hi = f16(w), lo = f16(w - hi)   (Markidis split)
// ---------------------------------------------------------------------------
__global__ __launch_bounds__(256)
void convert_hilo(const float* __restrict__ W,
                  _Float16* __restrict__ hi, _Float16* __restrict__ lo)
{
  long i = (long)blockIdx.x * 256 + threadIdx.x;
  float x = W[i];
  _Float16 h = (_Float16)x;
  hi[i] = h;
  lo[i] = (_Float16)(x - (float)h);
}

// ---------------------------------------------------------------------------
// Stage B: offset MLP -> (c0, c1, w0e, w1e) tables, index = ((b*L+l)*KS+ks)*H+h
// ---------------------------------------------------------------------------
__global__ __launch_bounds__(256)
void offset_mlp_kernel(const float* __restrict__ aux,
                       const float* __restrict__ Ow1, const float* __restrict__ Ob1,
                       const float* __restrict__ Ow2, const float* __restrict__ Ob2,
                       const float* __restrict__ Ow3,
                       int*   __restrict__ tc0, int*   __restrict__ tc1,
                       float* __restrict__ tw0, float* __restrict__ tw1)
{
  __shared__ float s1[128*64];   // Ow1[j][d]
  __shared__ float s2[128*64];   // Ow2 transposed: s2[j*64+d] = Ow2[d][j]
  __shared__ float sb1[128];
  __shared__ float sb2[64];
  __shared__ float s3[64];

  for (int e = threadIdx.x; e < 128*64; e += 256) s1[e] = Ow1[e];
  for (int e = threadIdx.x; e < 128*64; e += 256) {
    int j = e >> 6, d = e & 63;
    s2[e] = Ow2[d*128 + j];
  }
  for (int e = threadIdx.x; e < 128; e += 256) sb1[e] = Ob1[e];
  if (threadIdx.x < 64) { sb2[threadIdx.x] = Ob2[threadIdx.x];
                          s3[threadIdx.x]  = Ow3[threadIdx.x]; }
  __syncthreads();

  long gid = (long)blockIdx.x * 256 + threadIdx.x;       // (m,h) flat
  int  h  = (int)(gid & (HH-1));
  long m  = gid >> 4;                                    // (b,l,ks)
  int  ks = (int)(m % KSW);
  long g  = m / KSW;
  int  l  = (int)(g % LL);
  int  b  = (int)(g / LL);

  int left = l - KSW/2;
  if (left < 0) left = 0;
  if (left > LL - KSW) left = LL - KSW;
  int iw = left + ks;

  const float* arow = aux + ((long)b*LL + iw) * EE + h*DD;
  float a[64];
#pragma unroll
  for (int d = 0; d < 64; d += 4) {
    float4 t = *(const float4*)(arow + d);
    a[d] = t.x; a[d+1] = t.y; a[d+2] = t.z; a[d+3] = t.w;
  }

  float acc2[64];
#pragma unroll
  for (int d = 0; d < 64; ++d) acc2[d] = sb2[d];

  for (int j = 0; j < 128; ++j) {
    const float* w1r = s1 + j*64;
    float h1 = sb1[j];
#pragma unroll
    for (int d = 0; d < 64; ++d) h1 = fmaf(a[d], w1r[d], h1);
    h1 = fmaxf(h1, 0.0f);
    const float* w2c = s2 + j*64;
#pragma unroll
    for (int d = 0; d < 64; ++d) acc2[d] = fmaf(h1, w2c[d], acc2[d]);
  }

  float off = 0.0f;
#pragma unroll
  for (int d = 0; d < 64; ++d) off = fmaf(fmaxf(acc2[d], 0.0f), s3[d], off);

  float pos = (float)iw + off;
  float x0  = floorf(pos);
  float w1f = pos - x0;
  int i0 = (int)x0;
  int i1 = i0 + 1;
  float v0 = (i0 >= 0 && i0 < LL) ? 1.0f : 0.0f;
  float v1 = (i1 >= 0 && i1 < LL) ? 1.0f : 0.0f;
  int c0 = min(max(i0, 0), LL-1);
  int c1 = min(max(i1, 0), LL-1);

  tc0[gid] = c0;
  tc1[gid] = c1;
  tw0[gid] = (1.0f - w1f) * v0;
  tw1[gid] = w1f * v1;
}

// ---------------------------------------------------------------------------
// Dense exact-fp32 GEMM: Y[M,1024] = X[M,1024] * W[1024,1024]^T + bias
// wave tile 32x64 (2x4 wmma accumulators), block = 8 waves = 64M x 256N
// ---------------------------------------------------------------------------
__global__ __launch_bounds__(256)
void gemm_xwT_bias(const float* __restrict__ X, const float* __restrict__ W,
                   const float* __restrict__ bias, float* __restrict__ Y)
{
  const int K = EE, N = EE;
  int lane = threadIdx.x & 31;
  int wave = threadIdx.x >> 5;
  int wm = wave >> 2, wn = wave & 3;
  int r  = lane & 15, kg = lane >> 4;

  long m0 = (long)blockIdx.x * 64 + wm*32;
  int  n0 = blockIdx.y * 256 + wn*64;

  const float* x0p = X + (m0 + r)      * K + 2*kg;
  const float* x1p = X + (m0 + 16 + r) * K + 2*kg;
  const float* wp[4];
#pragma unroll
  for (int t = 0; t < 4; ++t) wp[t] = W + (long)(n0 + 16*t + r) * K + 2*kg;

  v8f acc[2][4] = {};
  for (int k0 = 0; k0 < K; k0 += 4) {
    v2f a0 = *(const v2f*)(x0p + k0);
    v2f a1 = *(const v2f*)(x1p + k0);
#pragma unroll
    for (int t = 0; t < 4; ++t) {
      v2f bt = *(const v2f*)(wp[t] + k0);
      acc[0][t] = wmma_f32(a0, bt, acc[0][t]);
      acc[1][t] = wmma_f32(a1, bt, acc[1][t]);
    }
  }

#pragma unroll
  for (int i = 0; i < 2; ++i)
#pragma unroll
    for (int t = 0; t < 4; ++t)
#pragma unroll
      for (int rr = 0; rr < 8; ++rr) {
        long m = m0 + 16*i + rr + 8*kg;
        int  n = n0 + 16*t + r;
        Y[m*(long)N + n] = acc[i][t][rr] + bias[n];
      }
}

// ---------------------------------------------------------------------------
// Fused deformable-sample + projection GEMM on the f16 WMMA pipe with
// split-precision (hi/lo) error compensation. The per-block interpolation
// tables are staged into LDS with a single TDM tensor_load_to_lds (2D tile:
// 1024 elems x 4 rows, row stride = NTBL), tracked by TENSORcnt.
// ---------------------------------------------------------------------------
__global__ __launch_bounds__(256)
void gemm_sampled_split16(const float* __restrict__ SRC,
                          const _Float16* __restrict__ Whi,
                          const _Float16* __restrict__ Wlo,
                          const float* __restrict__ bias,
                          const int* __restrict__ tc0,   // tc1/tw0/tw1 follow at +NTBL
                          float* __restrict__ Y, long mbase)
{
  const int K = EE, N = EE;
  __shared__ int stbl[4*1024];       // rows: tc0 | tc1 | tw0 | tw1 (block slice)
  int*   sc0 = stbl;
  int*   sc1 = stbl + 1024;
  float* sw0 = (float*)(stbl + 2048);
  float* sw1 = (float*)(stbl + 3072);

  long mg0 = mbase + (long)blockIdx.x * 64;

  if (threadIdx.x < 32) {            // one wave issues the DMA
    unsigned lds_off = (unsigned)(uintptr_t)&stbl[0];
    tdm_load_2d_b32(lds_off, tc0 + mg0*HH, 1024u, 4u,
                    (unsigned long long)NTBL_C);
    __builtin_amdgcn_s_wait_tensorcnt(0);
  }
  __syncthreads();

  int lane = threadIdx.x & 31;
  int wave = threadIdx.x >> 5;
  int wm = wave >> 2, wn = wave & 3;
  int r  = lane & 15, kg = lane >> 4;
  int n0 = blockIdx.y * 256 + wn*64;

  int b = (int)(mg0 / ((long)LL * KSW));
  const float* srcb = SRC + (long)b * LL * EE;

  int mloc[2];
  mloc[0] = wm*32 + r;
  mloc[1] = wm*32 + 16 + r;

  const _Float16* whp[4];
  const _Float16* wlp[4];
#pragma unroll
  for (int t = 0; t < 4; ++t) {
    long row = (long)(n0 + 16*t + r) * K + 8*kg;
    whp[t] = Whi + row;
    wlp[t] = Wlo + row;
  }

  v8f acc[2][4] = {};

  for (int k0 = 0; k0 < K; k0 += 32) {
    // ---- build hi/lo A operands (sampled in fp32, split to f16) ----
    v16h ahi[2], alo[2];
#pragma unroll
    for (int i = 0; i < 2; ++i) {
      int ml = mloc[i];
      float av[16];
#pragma unroll
      for (int runi = 0; runi < 2; ++runi) {
        int kk = k0 + 16*runi + 8*kg;          // 8 contiguous k, one head
        int h  = kk >> 6;
        float w0 = sw0[ml*16 + h], w1 = sw1[ml*16 + h];
        const float* p0 = srcb + (long)sc0[ml*16 + h]*EE + kk;
        const float* p1 = srcb + (long)sc1[ml*16 + h]*EE + kk;
        float4 a0 = *(const float4*)(p0);
        float4 a1 = *(const float4*)(p0 + 4);
        float4 b0 = *(const float4*)(p1);
        float4 b1 = *(const float4*)(p1 + 4);
        float* o = av + 8*runi;
        o[0] = w0*a0.x + w1*b0.x;  o[1] = w0*a0.y + w1*b0.y;
        o[2] = w0*a0.z + w1*b0.z;  o[3] = w0*a0.w + w1*b0.w;
        o[4] = w0*a1.x + w1*b1.x;  o[5] = w0*a1.y + w1*b1.y;
        o[6] = w0*a1.z + w1*b1.z;  o[7] = w0*a1.w + w1*b1.w;
      }
#pragma unroll
      for (int e = 0; e < 16; ++e) {
        float x = av[e];
        _Float16 hx = (_Float16)x;
        ahi[i][e] = hx;
        alo[i][e] = (_Float16)(x - (float)hx);
      }
    }
    // ---- B operands: two runs of 8 pre-split halves per lane (k=8kg.., 16+8kg..) ----
#pragma unroll
    for (int t = 0; t < 4; ++t) {
      v16h bhi, blo;
      {
        const float4 h0 = *(const float4*)(whp[t] + k0);
        const float4 h1 = *(const float4*)(whp[t] + k0 + 16);
        ((float4*)&bhi)[0] = h0;  ((float4*)&bhi)[1] = h1;
        const float4 l0 = *(const float4*)(wlp[t] + k0);
        const float4 l1 = *(const float4*)(wlp[t] + k0 + 16);
        ((float4*)&blo)[0] = l0;  ((float4*)&blo)[1] = l1;
      }
#pragma unroll
      for (int i = 0; i < 2; ++i) {
        acc[i][t] = wmma_f16(ahi[i], bhi, acc[i][t]);
        acc[i][t] = wmma_f16(alo[i], bhi, acc[i][t]);
        acc[i][t] = wmma_f16(ahi[i], blo, acc[i][t]);
      }
    }
  }

#pragma unroll
  for (int i = 0; i < 2; ++i)
#pragma unroll
    for (int t = 0; t < 4; ++t)
#pragma unroll
      for (int rr = 0; rr < 8; ++rr) {
        long m = (long)blockIdx.x*64 + wm*32 + 16*i + rr + 8*kg;  // chunk-local
        int  n = n0 + 16*t + r;
        Y[m*(long)N + n] = acc[i][t][rr] + bias[n];
      }
}

// ---------------------------------------------------------------------------
// Attention: one wave per (b,l,h); KS=13, D=64 (2 components per lane)
// ---------------------------------------------------------------------------
__global__ __launch_bounds__(256)
void attention_kernel(const float* __restrict__ qbuf,
                      const float* __restrict__ kbuf,  // chunk-local [GPC*KS][E]
                      const float* __restrict__ vbuf,
                      float* __restrict__ attn, long g0)
{
  int lane = threadIdx.x & 31;
  int wave = threadIdx.x >> 5;
  long wid = (long)blockIdx.x * 8 + wave;
  int  h  = (int)(wid & (HH-1));
  long gl = wid >> 4;                 // local (b,l) in chunk
  long g  = g0 + gl;                  // global b*L + l
  int  col = h*DD + 2*lane;

  v2f q = *(const v2f*)(qbuf + g*EE + col);
  const float scale = 0.125f;         // 1/sqrt(64)

  float logits[KSW];
#pragma unroll
  for (int ks = 0; ks < KSW; ++ks) {
    v2f kk = *(const v2f*)(kbuf + (gl*KSW + ks)*(long)EE + col);
    float p = q.x*kk.x + q.y*kk.y;
#pragma unroll
    for (int s = 16; s > 0; s >>= 1) p += __shfl_xor(p, s, 32);
    logits[ks] = p * scale;
  }

  float mx = logits[0];
#pragma unroll
  for (int ks = 1; ks < KSW; ++ks) mx = fmaxf(mx, logits[ks]);
  float aw[KSW], den = 0.0f;
#pragma unroll
  for (int ks = 0; ks < KSW; ++ks) { aw[ks] = __expf(logits[ks] - mx); den += aw[ks]; }
  float inv = 1.0f / den;

  v2f o = {0.0f, 0.0f};
#pragma unroll
  for (int ks = 0; ks < KSW; ++ks) {
    v2f vv = *(const v2f*)(vbuf + (gl*KSW + ks)*(long)EE + col);
    o += vv * (aw[ks] * inv);
  }
  *(v2f*)(attn + g*EE + col) = o;
}

// ---------------------------------------------------------------------------
// Host launcher
// ---------------------------------------------------------------------------
extern "C" void kernel_launch(void* const* d_in, const int* in_sizes, int n_in,
                              void* d_out, int out_size, void* d_ws, size_t ws_size,
                              hipStream_t stream) {
  const float* queries = (const float*)d_in[0];
  const float* keys    = (const float*)d_in[1];
  const float* values  = (const float*)d_in[2];
  const float* aux     = (const float*)d_in[3];
  const float* Wq = (const float*)d_in[4];  const float* bq = (const float*)d_in[5];
  const float* Wk = (const float*)d_in[6];  const float* bk = (const float*)d_in[7];
  const float* Wv = (const float*)d_in[8];  const float* bv = (const float*)d_in[9];
  const float* Wo = (const float*)d_in[10]; const float* bo = (const float*)d_in[11];
  const float* Ow1 = (const float*)d_in[12]; const float* Ob1 = (const float*)d_in[13];
  const float* Ow2 = (const float*)d_in[14]; const float* Ob2 = (const float*)d_in[15];
  const float* Ow3 = (const float*)d_in[16];
  float* out = (float*)d_out;

  // workspace layout (floats)
  const size_t NQ   = (size_t)BB*LL*EE;          // 4,194,304
  const size_t NTBL = NTBL_C;                    // 851,968
  const size_t NKB  = (size_t)MPC*EE;            // 6,815,744 (per chunk)
  const size_t NW   = (size_t)EE*EE;             // 1,048,576 weights
  float* ws = (float*)d_ws;
  float* qbuf = ws;                // NQ
  float* attn = qbuf + NQ;         // NQ
  int*   tc0  = (int*)(attn + NQ); // NTBL  (tc1/tw0/tw1 must stay contiguous:
  int*   tc1  = tc0 + NTBL;        //        the TDM 2D tile strides by NTBL)
  float* tw0  = (float*)(tc1 + NTBL);
  float* tw1  = tw0 + NTBL;
  float* kbuf = tw1 + NTBL;        // NKB
  float* vbuf = kbuf + NKB;        // NKB
  _Float16* wkhi = (_Float16*)(vbuf + NKB);   // NW halves each
  _Float16* wklo = wkhi + NW;
  _Float16* wvhi = wklo + NW;
  _Float16* wvlo = wvhi + NW;      // total ~110 MB

  // Weight hi/lo pre-split for the f16 matrix pipe
  convert_hilo<<<dim3((unsigned)(NW/256)), 256, 0, stream>>>(Wk, wkhi, wklo);
  convert_hilo<<<dim3((unsigned)(NW/256)), 256, 0, stream>>>(Wv, wvhi, wvlo);

  // Stage B: offset MLP tables (851,968 threads)
  offset_mlp_kernel<<<dim3((unsigned)(NTBL/256)), 256, 0, stream>>>(
      aux, Ow1, Ob1, Ow2, Ob2, Ow3, tc0, tc1, tw0, tw1);

  // Stage A: q = queries @ Wq^T + bq   (M=4096, exact fp32 WMMA)
  gemm_xwT_bias<<<dim3(BB*LL/64, EE/256), 256, 0, stream>>>(queries, Wq, bq, qbuf);

  // Stages C+D, chunked over (b,l): sampled K/V projections on the f16 pipe
  for (int c = 0; c < NCHUNK; ++c) {
    long mbase = (long)c * MPC;
    gemm_sampled_split16<<<dim3(MPC/64, EE/256), 256, 0, stream>>>(
        keys, wkhi, wklo, bk, tc0, kbuf, mbase);
    gemm_sampled_split16<<<dim3(MPC/64, EE/256), 256, 0, stream>>>(
        values, wvhi, wvlo, bv, tc0, vbuf, mbase);
    attention_kernel<<<dim3(GPC*HH/8), 256, 0, stream>>>(
        qbuf, kbuf, vbuf, attn, (long)c * GPC);
  }

  // Stage E: out = attn @ Wo^T + bo (exact fp32 WMMA)
  gemm_xwT_bias<<<dim3(BB*LL/64, EE/256), 256, 0, stream>>>(attn, Wo, bo, out);
}
